// AttentionAIC_45397804319388
// MI455X (gfx1250) — matrix operand order repaired and checked by reading, not compiled, not run on hardware
//
#include <hip/hip_runtime.h>
#include <hip/hip_bf16.h>

typedef __bf16 bf16_t;
typedef __attribute__((ext_vector_type(16))) __bf16 v16bf;
typedef __attribute__((ext_vector_type(8)))  float   v8f;

#define D_MODEL 1536
#define S_HID   2048
#define S_ENC   256
#define T_ALL   2304
#define N_HEAD  24
#define HEAD_D  64

union ABReg { v16bf v; uint4 q[2]; };

__device__ __forceinline__ v8f wmma_bf16f32(const ABReg& a, const ABReg& b, v8f c) {
  // D(f32 16x16) = A(16x32 bf16) * B(32x16 bf16) + C
  return __builtin_amdgcn_wmma_f32_16x16x32_bf16(false, a.v, false, b.v, (short)0, c,
                                                 false, false);
}

// ---------------------------------------------------------------------------
// f32 -> bf16 elementwise convert (vectorized x4)
// ---------------------------------------------------------------------------
__global__ void cvt_f32_bf16(const float* __restrict__ in, bf16_t* __restrict__ out, int n4) {
  int i = blockIdx.x * blockDim.x + threadIdx.x;
  if (i >= n4) return;
  float4 f = reinterpret_cast<const float4*>(in)[i];
  union { bf16_t b[4]; uint2 u; } p;
  p.b[0] = (bf16_t)f.x; p.b[1] = (bf16_t)f.y;
  p.b[2] = (bf16_t)f.z; p.b[3] = (bf16_t)f.w;
  reinterpret_cast<uint2*>(out)[i] = p.u;
}

// ---------------------------------------------------------------------------
// Y[M x 1536] = X[M x 1536](bf16) @ W^T + bias, W is torch-layout [O][D] so
// B-operand element (k,n) = W[n*D + k] (contiguous in k along a W row).
// One wave computes a 16(M) x 64(N) tile; block = 4 waves = 4 M-tiles.
// A k order used here (16x32): lane half hh: elems i -> K=hh*16+i, M=lane&15
// B k order used here (32x16, same as A): lane half hh: elems i -> K=hh*16+i, N=lane&15
// C layout: vgpr r, lane: M = r + 8*(lane>>4), N = lane&15
// ---------------------------------------------------------------------------
__global__ __launch_bounds__(128) void gemm_bf16_f32(
    const bf16_t* __restrict__ X, const bf16_t* __restrict__ W,
    const float* __restrict__ bias, float* __restrict__ Y, int Mrows) {
  const int lane = threadIdx.x & 31;
  const int wid  = threadIdx.x >> 5;
  const int half = lane >> 4;
  const int l15  = lane & 15;
  const int tileM = blockIdx.y * 4 + wid;   // 16-row tile index
  const int tileN = blockIdx.x;             // 64-col tile index
  (void)Mrows;

  const bf16_t* xrow = X + (size_t)(tileM * 16 + l15) * D_MODEL;
  const v8f vzero = {0.f,0.f,0.f,0.f,0.f,0.f,0.f,0.f};
  v8f acc[4] = {vzero, vzero, vzero, vzero};

  for (int kb = 0; kb < D_MODEL; kb += 32) {
    ABReg a;
    a.q[0] = *reinterpret_cast<const uint4*>(xrow + kb + half * 16);
    a.q[1] = *reinterpret_cast<const uint4*>(xrow + kb + half * 16 + 8);
#pragma unroll
    for (int j = 0; j < 4; ++j) {
      const bf16_t* wrow =
          W + (size_t)(tileN * 64 + j * 16 + l15) * D_MODEL + kb + half * 16;
      ABReg b;
      b.q[0] = *reinterpret_cast<const uint4*>(wrow);
      b.q[1] = *reinterpret_cast<const uint4*>(wrow + 8);
      acc[j] = wmma_bf16f32(a, b, acc[j]);
    }
  }
#pragma unroll
  for (int j = 0; j < 4; ++j) {
    const int n = tileN * 64 + j * 16 + l15;
    const float bv = bias[n];
#pragma unroll
    for (int r = 0; r < 8; ++r) {
      const int m = tileM * 16 + r + half * 8;
      Y[(size_t)m * D_MODEL + n] = acc[j][r] + bv;
    }
  }
}

// ---------------------------------------------------------------------------
// RMSNorm (mode 0) / plain (mode 1) + stream-concat + repack to bf16 attention
// layouts.  One wave handles one (head h, token t) row of 64 values.
// mode 0: out[h][Tall][64]      (Q and K layouts)
// mode 1: out[h][64][T_ALL]     (V transposed for the P@V B-operand)
// ---------------------------------------------------------------------------
__global__ __launch_bounds__(256) void pack_qkv(
    const float* __restrict__ Yh, const float* __restrict__ Ye,
    const float* __restrict__ wh, const float* __restrict__ we,
    bf16_t* __restrict__ out, int Th, int Tall, int mode) {
  const int lane = threadIdx.x & 31;
  const int wid  = threadIdx.x >> 5;
  const int idx  = blockIdx.x * 8 + wid;
  const int h = idx / Tall;
  const int t = idx - h * Tall;

  const float* src = (t < Th) ? (Yh + (size_t)t * D_MODEL)
                              : (Ye + (size_t)(t - Th) * D_MODEL);
  float2 x = *reinterpret_cast<const float2*>(src + h * 64 + lane * 2);
  float y0 = x.x, y1 = x.y;

  if (mode == 0) {
    float ssq = y0 * y0 + y1 * y1;
#pragma unroll
    for (int off = 16; off > 0; off >>= 1) ssq += __shfl_xor(ssq, off, 32);
    const float inv = rsqrtf(ssq * (1.0f / 64.0f) + 1e-6f);
    const float* w = (t < Th) ? wh : we;
    y0 *= inv * w[lane * 2];
    y1 *= inv * w[lane * 2 + 1];
    union { bf16_t b[2]; unsigned u; } p;
    p.b[0] = (bf16_t)y0; p.b[1] = (bf16_t)y1;
    reinterpret_cast<unsigned*>(out)[(size_t)(h * Tall + t) * 32 + lane] = p.u;
  } else {
    out[(size_t)(h * 64 + lane * 2) * T_ALL + t]     = (bf16_t)y0;
    out[(size_t)(h * 64 + lane * 2 + 1) * T_ALL + t] = (bf16_t)y1;
  }
}

// ---------------------------------------------------------------------------
// Flash attention, one wave = 16 queries of one head; 8 waves per block.
// Q  : [H][2048][64]  (A-operand reads, contiguous 16B chunks)
// K  : [H][2304][64]  (B-operand for Q@K^T: column n = key row of K)
// Vt : [H][64][2304]  (B-operand for P@V:  column n = hd row of Vt)
// O  : [2048][1536] bf16 (head h occupies columns h*64..h*64+63)
// ---------------------------------------------------------------------------
__global__ __launch_bounds__(256) void attn_kernel(
    const bf16_t* __restrict__ Q, const bf16_t* __restrict__ K,
    const bf16_t* __restrict__ Vt, const float* __restrict__ mask,
    bf16_t* __restrict__ O) {
  __shared__ alignas(16) bf16_t Pt[8][16][32];   // per-wave P tile (1 KB each)

  const int lane = threadIdx.x & 31;
  const int wid  = threadIdx.x >> 5;
  const int half = lane >> 4;
  const int l15  = lane & 15;
  const int h  = blockIdx.y;
  const int qb = (blockIdx.x * 8 + wid) * 16;

  // Load Q A-operands: aq0 covers K(hd)=0..31, aq1 covers 32..63
  const bf16_t* qrow = Q + ((size_t)h * S_HID + qb + l15) * HEAD_D;
  ABReg aq0, aq1;
  aq0.q[0] = *reinterpret_cast<const uint4*>(qrow + half * 16);
  aq0.q[1] = *reinterpret_cast<const uint4*>(qrow + half * 16 + 8);
  aq1.q[0] = *reinterpret_cast<const uint4*>(qrow + 32 + half * 16);
  aq1.q[1] = *reinterpret_cast<const uint4*>(qrow + 32 + half * 16 + 8);

  const v8f vzero = {0.f,0.f,0.f,0.f,0.f,0.f,0.f,0.f};
  v8f acc[4] = {vzero, vzero, vzero, vzero};
  float mi[8], li[8];
#pragma unroll
  for (int r = 0; r < 8; ++r) { mi[r] = -1e30f; li[r] = 0.f; }

  const float scale = 0.125f;   // 64^-0.5

  for (int kt = 0; kt < T_ALL; kt += 32) {
    // ---- scores for 32 keys: two 16x16 C tiles ----
    v8f cs[2] = {vzero, vzero};
#pragma unroll
    for (int ct = 0; ct < 2; ++ct) {
      const bf16_t* krow = K + ((size_t)h * T_ALL + kt + ct * 16 + l15) * HEAD_D;
      ABReg b0, b1;
      b0.q[0] = *reinterpret_cast<const uint4*>(krow + half * 16);
      b0.q[1] = *reinterpret_cast<const uint4*>(krow + half * 16 + 8);
      b1.q[0] = *reinterpret_cast<const uint4*>(krow + 32 + half * 16);
      b1.q[1] = *reinterpret_cast<const uint4*>(krow + 32 + half * 16 + 8);
      cs[ct] = wmma_bf16f32(aq0, b0, cs[ct]);
      cs[ct] = wmma_bf16f32(aq1, b1, cs[ct]);
    }
    const float mk0 = mask[kt + l15];
    const float mk1 = mask[kt + 16 + l15];

    // ---- online softmax (rows live in 16-lane halves of each C vgpr) ----
#pragma unroll
    for (int r = 0; r < 8; ++r) {
      float s0 = cs[0][r] * scale + mk0;
      float s1 = cs[1][r] * scale + mk1;
      float rm = fmaxf(s0, s1);
#pragma unroll
      for (int off = 8; off > 0; off >>= 1) rm = fmaxf(rm, __shfl_xor(rm, off, 32));
      const float mnew = fmaxf(mi[r], rm);
      const float corr = __expf(mi[r] - mnew);
      const float p0 = __expf(s0 - mnew);
      const float p1 = __expf(s1 - mnew);
      float rs = p0 + p1;
#pragma unroll
      for (int off = 8; off > 0; off >>= 1) rs += __shfl_xor(rs, off, 32);
      li[r] = li[r] * corr + rs;
      mi[r] = mnew;
      acc[0][r] *= corr; acc[1][r] *= corr; acc[2][r] *= corr; acc[3][r] *= corr;
      // scatter P into LDS in [m][k] order (C-layout -> A-layout transpose)
      Pt[wid][r + half * 8][l15]      = (bf16_t)p0;
      Pt[wid][r + half * 8][16 + l15] = (bf16_t)p1;
    }

    // ---- reload P as a 16x32 A-operand ----
    ABReg pa;
    pa.q[0] = *reinterpret_cast<const uint4*>(&Pt[wid][l15][half * 16]);
    pa.q[1] = *reinterpret_cast<const uint4*>(&Pt[wid][l15][half * 16 + 8]);

    // ---- out += P @ V over this 32-key slab, 4 hd-subtiles ----
#pragma unroll
    for (int j = 0; j < 4; ++j) {
      const bf16_t* vrow =
          Vt + ((size_t)h * HEAD_D + j * 16 + l15) * T_ALL + kt + half * 16;
      ABReg b;
      b.q[0] = *reinterpret_cast<const uint4*>(vrow);
      b.q[1] = *reinterpret_cast<const uint4*>(vrow + 8);
      acc[j] = wmma_bf16f32(pa, b, acc[j]);
    }
  }

  // ---- epilogue: normalize and store bf16 into O[q][h*64+hd] ----
#pragma unroll
  for (int j = 0; j < 4; ++j) {
#pragma unroll
    for (int r = 0; r < 8; ++r) {
      const float v = acc[j][r] / li[r];
      O[(size_t)(qb + r + half * 8) * D_MODEL + h * 64 + j * 16 + l15] = (bf16_t)v;
    }
  }
}

// ---------------------------------------------------------------------------
// Host-side orchestration
// ---------------------------------------------------------------------------
extern "C" void kernel_launch(void* const* d_in, const int* in_sizes, int n_in,
                              void* d_out, int out_size, void* d_ws, size_t ws_size,
                              hipStream_t stream) {
  (void)in_sizes; (void)n_in; (void)out_size; (void)ws_size;
  const float* x_h  = (const float*)d_in[0];
  const float* x_e  = (const float*)d_in[1];
  const float* mask = (const float*)d_in[2];
  const float* Wq   = (const float*)d_in[3];
  const float* bq   = (const float*)d_in[4];
  const float* Wk   = (const float*)d_in[5];
  const float* bk   = (const float*)d_in[6];
  const float* Wv   = (const float*)d_in[7];
  const float* bv   = (const float*)d_in[8];
  const float* nq   = (const float*)d_in[9];
  const float* nk   = (const float*)d_in[10];
  // d_in[11..12] add_q_proj      : dead (context attention output discarded)
  const float* Wka  = (const float*)d_in[13];
  const float* bka  = (const float*)d_in[14];
  const float* Wva  = (const float*)d_in[15];
  const float* bva  = (const float*)d_in[16];
  // d_in[17] norm_added_q        : dead
  const float* nka  = (const float*)d_in[18];
  const float* Wo   = (const float*)d_in[19];
  const float* bo   = (const float*)d_in[20];
  // d_in[21..22] to_add_out      : dead

  char* ws = (char*)d_ws;
  size_t off = 0;
  auto alloc = [&](size_t bytes) -> void* {
    void* p = ws + off;
    off = (off + bytes + 255) & ~(size_t)255;
    return p;
  };

  bf16_t* Xh   = (bf16_t*)alloc((size_t)S_HID * D_MODEL * 2);
  bf16_t* Xe   = (bf16_t*)alloc((size_t)S_ENC * D_MODEL * 2);
  bf16_t* bWq  = (bf16_t*)alloc((size_t)D_MODEL * D_MODEL * 2);
  bf16_t* bWk  = (bf16_t*)alloc((size_t)D_MODEL * D_MODEL * 2);
  bf16_t* bWv  = (bf16_t*)alloc((size_t)D_MODEL * D_MODEL * 2);
  bf16_t* bWka = (bf16_t*)alloc((size_t)D_MODEL * D_MODEL * 2);
  bf16_t* bWva = (bf16_t*)alloc((size_t)D_MODEL * D_MODEL * 2);
  bf16_t* bWo  = (bf16_t*)alloc((size_t)D_MODEL * D_MODEL * 2);
  float*  Yq   = (float*)alloc((size_t)S_HID * D_MODEL * 4);
  float*  Ykh  = (float*)alloc((size_t)S_HID * D_MODEL * 4);
  float*  Yvh  = (float*)alloc((size_t)S_HID * D_MODEL * 4);
  float*  Yke  = (float*)alloc((size_t)S_ENC * D_MODEL * 4);
  float*  Yve  = (float*)alloc((size_t)S_ENC * D_MODEL * 4);
  bf16_t* Qb   = (bf16_t*)alloc((size_t)N_HEAD * S_HID * HEAD_D * 2);
  bf16_t* Kb   = (bf16_t*)alloc((size_t)N_HEAD * T_ALL * HEAD_D * 2);
  bf16_t* Vtb  = (bf16_t*)alloc((size_t)N_HEAD * T_ALL * HEAD_D * 2);
  bf16_t* Ob   = (bf16_t*)alloc((size_t)S_HID * D_MODEL * 2);

  auto cvt = [&](const float* in, bf16_t* out, size_t n) {
    int n4 = (int)(n / 4);
    cvt_f32_bf16<<<(n4 + 255) / 256, 256, 0, stream>>>(in, out, n4);
  };
  cvt(x_h, Xh, (size_t)S_HID * D_MODEL);
  cvt(x_e, Xe, (size_t)S_ENC * D_MODEL);
  cvt(Wq,  bWq,  (size_t)D_MODEL * D_MODEL);
  cvt(Wk,  bWk,  (size_t)D_MODEL * D_MODEL);
  cvt(Wv,  bWv,  (size_t)D_MODEL * D_MODEL);
  cvt(Wka, bWka, (size_t)D_MODEL * D_MODEL);
  cvt(Wva, bWva, (size_t)D_MODEL * D_MODEL);
  cvt(Wo,  bWo,  (size_t)D_MODEL * D_MODEL);

  auto gemm = [&](const bf16_t* X, const bf16_t* W, const float* bias, float* Y, int M) {
    dim3 grid(D_MODEL / 64, M / 64);
    gemm_bf16_f32<<<grid, 128, 0, stream>>>(X, W, bias, Y, M);
  };
  gemm(Xh, bWq,  bq,  Yq,  S_HID);
  gemm(Xh, bWk,  bk,  Ykh, S_HID);
  gemm(Xh, bWv,  bv,  Yvh, S_HID);
  gemm(Xe, bWka, bka, Yke, S_ENC);
  gemm(Xe, bWva, bva, Yve, S_ENC);

  pack_qkv<<<(N_HEAD * S_HID) / 8, 256, 0, stream>>>(Yq, Yq, nq, nq, Qb,
                                                     S_HID, S_HID, 0);
  pack_qkv<<<(N_HEAD * T_ALL) / 8, 256, 0, stream>>>(Ykh, Yke, nk, nka, Kb,
                                                     S_HID, T_ALL, 0);
  pack_qkv<<<(N_HEAD * T_ALL) / 8, 256, 0, stream>>>(Yvh, Yve, nullptr, nullptr, Vtb,
                                                     S_HID, T_ALL, 1);

  dim3 ga(S_HID / 128, N_HEAD);
  attn_kernel<<<ga, 256, 0, stream>>>(Qb, Kb, Vtb, mask, Ob);

  // final projection straight into d_out (f32 + bias)
  gemm(Ob, bWo, bo, (float*)d_out, S_HID);
}
